// DKVMN_41704132444428
// MI455X (gfx1250) — matrix operand order repaired and checked by reading, not compiled, not run on hardware
//
#include <hip/hip_runtime.h>
#include <math.h>

typedef _Float16 half_t;
typedef __attribute__((ext_vector_type(16))) _Float16 v16h;
typedef __attribute__((ext_vector_type(8)))  float    v8f;
typedef __attribute__((ext_vector_type(4)))  float    v4f;
typedef __attribute__((ext_vector_type(4)))  unsigned int v4u;

#define B_   256
#define T_   1024
#define KD_  50
#define VD_  200
#define M_   20
#define FC_  50
#define ROWS (B_ * T_)        /* 262144 */
#define RTILES (ROWS / 16)    /* 16384  */

/* erase/add GEMM: K=200 -> 7 k-tiles of 32; N=200 -> 13 col tiles of 16 */
#define EA_KT 7
#define EA_NT 13
/* read GEMM: K=256 (200 reads + 50 q + 6 zero) -> 8 k-tiles; N=64 -> 4 col tiles */
#define RD_KT 8
#define RD_NT 4
#define PINLD 256             /* padded predict-input row length (halves) */

union frag16 { v16h h; v4u u[2]; half_t e[16]; };

__device__ __forceinline__ v8f wmma_f16(v16h a, v16h b, v8f c) {
  return __builtin_amdgcn_wmma_f32_16x16x32_f16(false, a, false, b, (short)0, c,
                                                false, false);
}

/* Load one 16x32 A-fragment from an f32 row (this lane owns row m=lane&15).
 * Per ISA 7.12.2: lane<16 half h -> K = h + 8*(h>=8); lane>=16 adds +8.
 * With k0 = kBase + 8*(lane>=16): halves 0..7 <- k0..k0+7, halves 8..15 <- k0+16..k0+23. */
__device__ __forceinline__ v16h load_a_f32(const float* __restrict__ row,
                                           int kBase, int khi, int Kmax) {
  frag16 f;
  int k0 = kBase + 8 * khi;
  if (k0 + 24 <= Kmax) {
    v4f x0 = *(const v4f*)(row + k0);
    v4f x1 = *(const v4f*)(row + k0 + 4);
    v4f x2 = *(const v4f*)(row + k0 + 16);
    v4f x3 = *(const v4f*)(row + k0 + 20);
#pragma unroll
    for (int i = 0; i < 4; ++i) {
      f.e[i]      = (half_t)x0[i];
      f.e[4 + i]  = (half_t)x1[i];
      f.e[8 + i]  = (half_t)x2[i];
      f.e[12 + i] = (half_t)x3[i];
    }
  } else {
#pragma unroll
    for (int h = 0; h < 16; ++h) {
      int k = k0 + (h & 7) + ((h >= 8) ? 16 : 0);
      f.e[h] = (k < Kmax) ? (half_t)row[k] : (half_t)0.0f;
    }
  }
  return f.h;
}

/* -------- weight pre-swizzle into B-fragment order ----------------------
 * out[((kt*NT+nt)*32 + lane)*16 + h] = W[j*wstride + k] (f16), zero padded,
 * with k = kt*32 + 8*(lane>=16) + (h&7) + 16*(h>=8), j = nt*16 + (lane&15). */
__global__ void prep_bsw_kernel(const float* __restrict__ W, int wstride,
                                int Jmax, int Kmax, int NT, int total,
                                half_t* __restrict__ outb) {
  int idx = blockIdx.x * blockDim.x + threadIdx.x;
  if (idx >= total) return;
  int h    = idx & 15;
  int lane = (idx >> 4) & 31;
  int tile = idx >> 9;
  int nt = tile % NT, kt = tile / NT;
  int k = kt * 32 + 8 * (lane >> 4) + (h & 7) + ((h >= 8) ? 16 : 0);
  int j = nt * 16 + (lane & 15);
  float v = (k < Kmax && j < Jmax) ? W[j * wstride + k] : 0.0f;
  outb[idx] = (half_t)v;
}

/* -------- fill predict-input columns 200..255 with f16(q) / zero -------- */
__global__ void pack_pinq_kernel(const float* __restrict__ q,
                                 half_t* __restrict__ pin) {
  size_t idx = (size_t)blockIdx.x * blockDim.x + threadIdx.x;
  if (idx >= (size_t)ROWS * 56) return;
  int c = (int)(idx % 56);
  size_t r = idx / 56;
  half_t v = (c < KD_) ? (half_t)q[r * KD_ + c] : (half_t)0.0f;
  pin[r * PINLD + VD_ + c] = v;
}

/* -------- addressing: w = softmax(q @ key_memory^T), rows = B*T --------- */
__global__ void addr_softmax_kernel(const float* __restrict__ q,
                                    const float* __restrict__ keymem,
                                    float* __restrict__ wout) {
  __shared__ float km[M_ * KD_];
  for (int i = threadIdx.x; i < M_ * KD_; i += blockDim.x) km[i] = keymem[i];
  __syncthreads();
  size_t r = (size_t)blockIdx.x * blockDim.x + threadIdx.x;
  if (r >= (size_t)ROWS) return;
  float qv[KD_];
#pragma unroll
  for (int k = 0; k < KD_; ++k) qv[k] = q[r * KD_ + k];
  float s[M_];
  float mx = -1e30f;
#pragma unroll
  for (int m = 0; m < M_; ++m) {
    float d = 0.0f;
#pragma unroll
    for (int k = 0; k < KD_; ++k) d += qv[k] * km[m * KD_ + k];
    s[m] = d;
    mx = fmaxf(mx, d);
  }
  float sum = 0.0f;
#pragma unroll
  for (int m = 0; m < M_; ++m) { s[m] = __expf(s[m] - mx); sum += s[m]; }
  float inv = 1.0f / sum;
#pragma unroll
  for (int m = 0; m < M_; ++m) wout[r * M_ + m] = s[m] * inv;
}

/* -------- erase/add GEMM: one wave per 16-row tile, A kept in VGPRs ----- */
__global__ __launch_bounds__(32) void gemm_ea_kernel(
    const float* __restrict__ qa, const half_t* __restrict__ bswE,
    const half_t* __restrict__ bswA, const float* __restrict__ biasE,
    const float* __restrict__ biasA, half_t* __restrict__ eraseO,
    half_t* __restrict__ addO) {
  int lane = threadIdx.x;
  int m = lane & 15, khi = lane >> 4;
  size_t rowBase = (size_t)blockIdx.x * 16;
  const float* arow = qa + (rowBase + m) * VD_;
  v16h afr[EA_KT];
#pragma unroll
  for (int kt = 0; kt < EA_KT; ++kt) afr[kt] = load_a_f32(arow, kt * 32, khi, VD_);

  for (int ct = 0; ct < EA_NT; ++ct) {
    v8f accE = {}; v8f accA = {};
#pragma unroll
    for (int kt = 0; kt < EA_KT; ++kt) {
      size_t tb = (((size_t)kt * EA_NT + ct) * 32 + lane) * 16;
      v16h be = *(const v16h*)(bswE + tb);
      v16h ba = *(const v16h*)(bswA + tb);
      accE = wmma_f16(afr[kt], be, accE);
      accA = wmma_f16(afr[kt], ba, accA);
    }
    int col = ct * 16 + (lane & 15);
    if (col < VD_) {
      float bE = biasE[col], bA = biasA[col];
#pragma unroll
      for (int r = 0; r < 8; ++r) {
        size_t row = rowBase + r + 8 * khi;
        float ev = 1.0f / (1.0f + __expf(-(accE[r] + bE)));
        float av = tanhf(accA[r] + bA);
        eraseO[row * VD_ + col] = (half_t)ev;
        addO[row * VD_ + col]   = (half_t)av;
      }
    }
  }
}

/* -------- sequential memory scan: 1 block / batch, 1 thread / VD dim ---- */
__global__ __launch_bounds__(256) void scan_kernel(
    const float* __restrict__ w, const half_t* __restrict__ eraseI,
    const half_t* __restrict__ addI, const float* __restrict__ mem0,
    half_t* __restrict__ pin) {
  const int b = blockIdx.x;
  const int d = threadIdx.x;
  float mem[M_];
  if (d < VD_) {
#pragma unroll
    for (int m = 0; m < M_; ++m) mem[m] = mem0[m * VD_ + d];
  }
  __shared__ float ws_s[8 * M_];
  for (int t0 = 0; t0 < T_; t0 += 8) {
    int i = threadIdx.x;
    if (i < 8 * M_)
      ws_s[i] = w[((size_t)b * T_ + t0 + i / M_) * M_ + (i % M_)];
    __syncthreads();
#pragma unroll
    for (int tt = 0; tt < 8; ++tt) {
      if (d < VD_) {
        size_t r = (size_t)b * T_ + t0 + tt;
        float e = (float)eraseI[r * VD_ + d];
        float a = (float)addI[r * VD_ + d];
        float rd = 0.0f;
#pragma unroll
        for (int m = 0; m < M_; ++m) {
          float wm = ws_s[tt * M_ + m];
          rd += wm * mem[m];                                /* read old mem */
          mem[m] = mem[m] * (1.0f - wm * e) + wm * a;       /* update       */
        }
        pin[r * PINLD + d] = (half_t)rd;
      }
    }
    __syncthreads();
  }
}

/* -------- final GEMM + tanh: pin (f16, K=256) x W_read^T -> f32 out ----- */
__global__ __launch_bounds__(32) void gemm_read_kernel(
    const half_t* __restrict__ pin, const half_t* __restrict__ bswR,
    const float* __restrict__ biasR, float* __restrict__ out) {
  int lane = threadIdx.x;
  int m = lane & 15, khi = lane >> 4;
  size_t rowBase = (size_t)blockIdx.x * 16;
  const half_t* arow = pin + (rowBase + m) * PINLD;
  v16h afr[RD_KT];
#pragma unroll
  for (int kt = 0; kt < RD_KT; ++kt) {
    frag16 f;
    int k0 = kt * 32 + 8 * khi;
    f.u[0] = *(const v4u*)(arow + k0);
    f.u[1] = *(const v4u*)(arow + k0 + 16);
    afr[kt] = f.h;
  }
#pragma unroll
  for (int ct = 0; ct < RD_NT; ++ct) {
    v8f acc = {};
#pragma unroll
    for (int kt = 0; kt < RD_KT; ++kt) {
      v16h bfr = *(const v16h*)(bswR + (((size_t)kt * RD_NT + ct) * 32 + lane) * 16);
      acc = wmma_f16(afr[kt], bfr, acc);
    }
    int col = ct * 16 + (lane & 15);
    if (col < FC_) {
      float bb = biasR[col];
#pragma unroll
      for (int r = 0; r < 8; ++r) {
        size_t row = rowBase + r + 8 * khi;
        out[row * FC_ + col] = tanhf(acc[r] + bb);
      }
    }
  }
}

extern "C" void kernel_launch(void* const* d_in, const int* in_sizes, int n_in,
                              void* d_out, int out_size, void* d_ws, size_t ws_size,
                              hipStream_t stream) {
  const float* q_embed  = (const float*)d_in[0];   /* (B,T,KD)  */
  const float* qa_embed = (const float*)d_in[1];   /* (B,T,VD)  */
  const float* key_mem  = (const float*)d_in[2];   /* (M,KD)    */
  const float* mem0     = (const float*)d_in[3];   /* (M,VD)    */
  const float* W_erase  = (const float*)d_in[4];   /* (VD,VD)   */
  const float* b_erase  = (const float*)d_in[5];
  const float* W_add    = (const float*)d_in[6];
  const float* b_add    = (const float*)d_in[7];
  const float* W_read   = (const float*)d_in[8];   /* (FC, VD+FC=250) */
  const float* b_read   = (const float*)d_in[9];
  float* out = (float*)d_out;                      /* (B,T,FC) f32 */

  /* workspace partition (all 256B aligned) */
  char* p = (char*)d_ws;
  auto take = [&p](size_t bytes) -> void* {
    void* r = (void*)p;
    p += (bytes + 255) & ~(size_t)255;
    return r;
  };
  float*  w_buf     = (float*) take((size_t)ROWS * M_ * sizeof(float));
  half_t* erase_buf = (half_t*)take((size_t)ROWS * VD_ * sizeof(half_t));
  half_t* add_buf   = (half_t*)take((size_t)ROWS * VD_ * sizeof(half_t));
  half_t* pin_buf   = (half_t*)take((size_t)ROWS * PINLD * sizeof(half_t));
  half_t* bswE      = (half_t*)take((size_t)EA_KT * EA_NT * 512 * sizeof(half_t));
  half_t* bswA      = (half_t*)take((size_t)EA_KT * EA_NT * 512 * sizeof(half_t));
  half_t* bswR      = (half_t*)take((size_t)RD_KT * RD_NT * 512 * sizeof(half_t));
  (void)ws_size; (void)in_sizes; (void)n_in; (void)out_size;

  /* 1) pre-swizzle weights into WMMA B-fragment order (f16) */
  {
    int totEA = EA_KT * EA_NT * 512;
    int totRD = RD_KT * RD_NT * 512;
    prep_bsw_kernel<<<(totEA + 255) / 256, 256, 0, stream>>>(
        W_erase, VD_, VD_, VD_, EA_NT, totEA, bswE);
    prep_bsw_kernel<<<(totEA + 255) / 256, 256, 0, stream>>>(
        W_add, VD_, VD_, VD_, EA_NT, totEA, bswA);
    prep_bsw_kernel<<<(totRD + 255) / 256, 256, 0, stream>>>(
        W_read, VD_ + FC_, FC_, VD_ + FC_, RD_NT, totRD, bswR);
  }

  /* 2) predict-input cols 200..255 = f16(q) / zero-pad */
  {
    size_t tot = (size_t)ROWS * 56;
    pack_pinq_kernel<<<(unsigned)((tot + 255) / 256), 256, 0, stream>>>(q_embed, pin_buf);
  }

  /* 3) addressing softmax (hoisted out of the scan) */
  addr_softmax_kernel<<<ROWS / 256, 256, 0, stream>>>(q_embed, key_mem, w_buf);

  /* 4) erase/add WMMA GEMMs (hoisted out of the scan) */
  gemm_ea_kernel<<<RTILES, 32, 0, stream>>>(qa_embed, bswE, bswA, b_erase, b_add,
                                            erase_buf, add_buf);

  /* 5) sequential memory scan; writes reads (f16) into predict-input */
  scan_kernel<<<B_, 256, 0, stream>>>(w_buf, erase_buf, add_buf, mem0, pin_buf);

  /* 6) final WMMA GEMM + tanh -> f32 output */
  gemm_read_kernel<<<RTILES, 32, 0, stream>>>(pin_buf, bswR, b_read, out);
}
